// HistNetworkModule_46780783788263
// MI455X (gfx1250) — compile-verified, matrix operand order checked by reading, not run on hardware
//
#include <hip/hip_runtime.h>
#include <hip/hip_bf16.h>

typedef __attribute__((ext_vector_type(16))) _Float16 v16h;
typedef __attribute__((ext_vector_type(8)))  float    v8f;

#define N_BAG   16384
#define DIN     128
#define FDIM    128
#define BINS    32
#define H1DIM   1024
#define H2DIM   512
#define NCLS    10
#define SIGMA_C 100.0f

#define ROWS_PER_WG 128
#define TPB1        128   // 4 waves of 32
#define LPAD        8     // +8 halves = 16B: keeps 16B alignment, bank stride 68%64=4

// ---------------------------------------------------------------------------
// Kernel 0: zero the histogram accumulator in workspace
// ---------------------------------------------------------------------------
__global__ void zero_hist(float* __restrict__ hist) {
    int i = blockIdx.x * blockDim.x + threadIdx.x;
    if (i < FDIM * BINS) hist[i] = 0.0f;
}

// ---------------------------------------------------------------------------
// Kernel 1: fused  feats = x@W_feat + b -> sigmoid -> soft-histogram partial
// WMMA f16 16x16x32; B stored K-major (transposed) in LDS so both A and B
// fragments load as two ds_load_b128 per WMMA. Histogram in registers.
// ---------------------------------------------------------------------------
__global__ __launch_bounds__(TPB1) void fused_feat_hist(
    const float* __restrict__ x,      // [N_BAG][DIN]
    const float* __restrict__ Wf,     // [DIN][FDIM]
    const float* __restrict__ bf,     // [FDIM]
    float* __restrict__ hist)         // [FDIM*BINS]  (pre-zeroed, atomics)
{
    __shared__ _Float16 sWt[FDIM][DIN + LPAD];  // ~34 KB  W_feat^T as f16: [n][k]
    __shared__ _Float16 sA[16][DIN + LPAD];     // ~4.3 KB x row-tile as f16: [m][k]
    __shared__ float    sS[16][FDIM];           //  8 KB   sigmoid(feats) tile

    const int t    = threadIdx.x;
    const int wave = t >> 5;
    const int lane = t & 31;
    const int hl   = lane & 15;   // half-lane: row (A) / col (B,C,D)
    const int hi   = lane >> 4;   // which 16-lane half

    // Stage W_feat -> LDS as f16, transposed to K-major (once per block).
    // Global reads coalesced; LDS write conflicts ~2-way due to LPAD.
    for (int idx = t; idx < DIN * FDIM; idx += TPB1) {
        int k = idx >> 7, n = idx & 127;
        sWt[n][k] = (_Float16)Wf[idx];
    }

    // Private per-column histogram accumulators
    float hacc[BINS];
#pragma unroll
    for (int k = 0; k < BINS; ++k) hacc[k] = 0.0f;

    const int row0 = blockIdx.x * ROWS_PER_WG;

    for (int rt = 0; rt < ROWS_PER_WG / 16; ++rt) {
        const int r0 = row0 + rt * 16;
        __syncthreads();   // sWt ready (iter 0); prev histogram read of sS done
        // Stage 16xDIN tile of x as f16
        for (int idx = t; idx < 16 * DIN; idx += TPB1) {
            int r = idx >> 7, c = idx & 127;
            sA[r][c] = (_Float16)x[(size_t)(r0 + r) * DIN + c];
        }
        // Prefetch next row-tile (8 KB = 64 x 128B lines) while we compute.
        if (rt + 1 < ROWS_PER_WG / 16 && t < 64) {
            __builtin_prefetch(&x[(size_t)(r0 + 16) * DIN + (size_t)t * 32], 0, 3);
        }
        __syncthreads();

        // Each wave: two 16x16 output tiles (covers all 128 feature cols)
        for (int ct = 0; ct < 2; ++ct) {
            const int n0   = wave * 32 + ct * 16;
            const int ncol = n0 + hl;
            v8f acc = {};
#pragma unroll
            for (int kt = 0; kt < 4; ++kt) {
                const int k0 = kt * 32;
                v16h a, b;
                // ISA 7.12.2 f16 A(16x32)/B(32x16) layouts:
                // lanes 0-15: K = 0..7 (v0..3), 16..23 (v4..7)
                // lanes 16-31: K = 8..15, 24..31 ; element pairs per VGPR.
                // Both fragments are K-contiguous in LDS -> 2x ds_load_b128 each.
#pragma unroll
                for (int j = 0; j < 8; ++j) {
                    a[j]     = sA[hl][k0 + hi * 8 + j];
                    a[8 + j] = sA[hl][k0 + 16 + hi * 8 + j];
                    b[j]     = sWt[ncol][k0 + hi * 8 + j];
                    b[8 + j] = sWt[ncol][k0 + 16 + hi * 8 + j];
                }
                acc = __builtin_amdgcn_wmma_f32_16x16x32_f16(
                        false, a, false, b, (short)0, acc, false, false);
            }
            // C/D layout: VGPR i -> M = i + 8*hi, N = hl  (same col for all 8)
            const float bias = bf[ncol];
#pragma unroll
            for (int i = 0; i < 8; ++i) {
                float z = acc[i] + bias;
                float s = 1.0f / (1.0f + __expf(-z));
                sS[i + hi * 8][ncol] = s;
            }
        }
        __syncthreads();

        // Histogram: thread t owns feature column t, folds 16 rows.
        // soft(k) = sigmoid(S*(s-k*d)) - sigmoid(S*(s-(k+1)*d)) : telescoping
#pragma unroll 2
        for (int r = 0; r < 16; ++r) {
            const float s = sS[r][t];
            float prev = 1.0f / (1.0f + __expf(-SIGMA_C * s));
#pragma unroll
            for (int k = 1; k <= BINS; ++k) {
                float cur = 1.0f / (1.0f + __expf(-SIGMA_C * (s - (float)k * (1.0f / BINS))));
                hacc[k - 1] += prev - cur;
                prev = cur;
            }
        }
    }

    const float inv_n = 1.0f / (float)N_BAG;
#pragma unroll
    for (int k = 0; k < BINS; ++k)
        atomicAdd(&hist[t * BINS + k], hacc[k] * inv_n);
}

// ---------------------------------------------------------------------------
// Kernel 2/3: GEMV + LeakyReLU(0.01). Coalesced along out_dim; vin broadcast.
// ---------------------------------------------------------------------------
__global__ __launch_bounds__(256) void gemv_lrelu(
    const float* __restrict__ vin, const float* __restrict__ W,
    const float* __restrict__ bvec, float* __restrict__ vout,
    int in_dim, int out_dim)
{
    int j = blockIdx.x * 256 + threadIdx.x;
    if (j >= out_dim) return;
    float acc = bvec[j];
    for (int i = 0; i < in_dim; ++i)
        acc = fmaf(vin[i], W[(size_t)i * out_dim + j], acc);
    vout[j] = acc > 0.0f ? acc : 0.01f * acc;
}

// ---------------------------------------------------------------------------
// Kernel 4: logits = h2@W3 + b3 ; softmax over 10 classes
// ---------------------------------------------------------------------------
__global__ __launch_bounds__(256) void head_softmax(
    const float* __restrict__ h2, const float* __restrict__ W3,
    const float* __restrict__ b3, float* __restrict__ out)
{
    __shared__ float red[256];
    __shared__ float logits[NCLS];
    const int t = threadIdx.x;
    for (int c = 0; c < NCLS; ++c) {
        float p = h2[t] * W3[t * NCLS + c]
                + h2[t + 256] * W3[(t + 256) * NCLS + c];
        red[t] = p;
        __syncthreads();
        for (int s = 128; s > 0; s >>= 1) {
            if (t < s) red[t] += red[t + s];
            __syncthreads();
        }
        if (t == 0) logits[c] = red[0] + b3[c];
        __syncthreads();
    }
    if (t == 0) {
        float m = logits[0];
        for (int c = 1; c < NCLS; ++c) m = fmaxf(m, logits[c]);
        float sum = 0.0f;
        float e[NCLS];
        for (int c = 0; c < NCLS; ++c) { e[c] = __expf(logits[c] - m); sum += e[c]; }
        for (int c = 0; c < NCLS; ++c) out[c] = e[c] / sum;
    }
}

// ---------------------------------------------------------------------------
extern "C" void kernel_launch(void* const* d_in, const int* in_sizes, int n_in,
                              void* d_out, int out_size, void* d_ws, size_t ws_size,
                              hipStream_t stream) {
    const float* x     = (const float*)d_in[0];
    const float* Wfeat = (const float*)d_in[1];
    const float* bfeat = (const float*)d_in[2];
    const float* W1    = (const float*)d_in[3];
    const float* b1    = (const float*)d_in[4];
    const float* W2    = (const float*)d_in[5];
    const float* b2    = (const float*)d_in[6];
    const float* W3    = (const float*)d_in[7];
    const float* b3    = (const float*)d_in[8];
    float* out = (float*)d_out;

    float* ws   = (float*)d_ws;
    float* hist = ws;                       // FDIM*BINS = 4096
    float* h1   = ws + FDIM * BINS;         // 1024
    float* h2   = h1 + H1DIM;               // 512

    zero_hist<<<(FDIM * BINS + 255) / 256, 256, 0, stream>>>(hist);

    fused_feat_hist<<<N_BAG / ROWS_PER_WG, TPB1, 0, stream>>>(x, Wfeat, bfeat, hist);

    gemv_lrelu<<<H1DIM / 256, 256, 0, stream>>>(hist, W1, b1, h1, FDIM * BINS, H1DIM);
    gemv_lrelu<<<H2DIM / 256, 256, 0, stream>>>(h1,   W2, b2, h2, H1DIM,       H2DIM);

    head_softmax<<<1, 256, 0, stream>>>(h2, W3, b3, out);
}
